// CasualCrossAttention_17987323036468
// MI455X (gfx1250) — matrix operand order repaired and checked by reading, not compiled, not run on hardware
//
#include <hip/hip_runtime.h>
#include <cstdint>

#define B_ 2
#define T_ 2048
#define S_ 2048
#define C_ 1024
#define H_ 16
#define D_ 64

typedef __attribute__((ext_vector_type(16))) __bf16 v16bf;
typedef __attribute__((ext_vector_type(8)))  float  v8f;
typedef __attribute__((ext_vector_type(4)))  unsigned int v4u;
typedef __attribute__((ext_vector_type(8)))  int    v8i;
typedef __attribute__((ext_vector_type(4)))  int    v4i;

typedef __attribute__((address_space(3))) void las_void;

#if __has_builtin(__builtin_amdgcn_global_load_async_to_lds_b128) && \
    __has_builtin(__builtin_amdgcn_s_wait_asynccnt)
#define HAVE_ASYNC 1
#else
#define HAVE_ASYNC 0
#endif

#if __has_builtin(__builtin_amdgcn_tensor_load_to_lds) && \
    __has_builtin(__builtin_amdgcn_s_wait_tensorcnt)
#define HAVE_TDM 1
#else
#define HAVE_TDM 0
#endif

// Contiguous k order used here for 16-bit A (and B) fragments: dword j of a lane in half `half`
// holds K values {kmapA, kmapA+1}.
__device__ __forceinline__ int kmapA(int j, int half) {
  return half * 16 + 2 * j;
}

// --- DPP16 butterfly reductions over each 16-lane row (wave32) -------------
// Replaces ds_bpermute-based __shfl_xor: pure VALU, no LDS traffic.
template <int CTRL>
__device__ __forceinline__ float dpp_mov(float v) {
  return __int_as_float(__builtin_amdgcn_update_dpp(
      0, __float_as_int(v), CTRL, 0xF, 0xF, true));
}
__device__ __forceinline__ float red16_max(float v) {
  v = fmaxf(v, dpp_mov<0xB1>(v));   // quad_perm(1,0,3,2)
  v = fmaxf(v, dpp_mov<0x4E>(v));   // quad_perm(2,3,0,1)
  v = fmaxf(v, dpp_mov<0x141>(v));  // row_half_mirror
  v = fmaxf(v, dpp_mov<0x140>(v));  // row_mirror
  return v;
}
__device__ __forceinline__ float red16_sum(float v) {
  v += dpp_mov<0xB1>(v);
  v += dpp_mov<0x4E>(v);
  v += dpp_mov<0x141>(v);
  v += dpp_mov<0x140>(v);
  return v;
}

// --- gfx1250 async global->LDS copy (16B per lane), sync fallback ----------
__device__ __forceinline__ void copy16_async(void* lds_dst, const void* gsrc) {
#if HAVE_ASYNC
  __builtin_amdgcn_global_load_async_to_lds_b128((v4i*)gsrc, (v4i*)lds_dst, 0, 0);
#else
  *reinterpret_cast<uint4*>(lds_dst) = *reinterpret_cast<const uint4*>(gsrc);
#endif
}
__device__ __forceinline__ void wait_async() {
#if HAVE_ASYNC
  __builtin_amdgcn_s_wait_asynccnt(0);
#endif
}

// --- Tensor Data Mover: 2D bf16 tile (tile_w x tile_h) into padded LDS -----
// Pads LDS dest by 4 dwords (8 bf16) after every 32 dwords (64 bf16) ->
// matches row stride of 72 bf16 for a 64-wide tile.
__device__ __forceinline__ void tdm_load_tile_2d(void* lds_dst, const void* gsrc,
                                                 int tile_w, int tile_h,
                                                 int row_stride_elems) {
#if HAVE_TDM
  unsigned lds_off = (unsigned)(uintptr_t)(las_void*)lds_dst;
  unsigned long long ga = (unsigned long long)(uintptr_t)gsrc;
  union { unsigned long long q[2]; v4u v; } g0;
  union { unsigned long long q[4]; v8i v; } g1;
  g0.q[0] = 1ull                                     // count = 1 descriptor
          | ((unsigned long long)lds_off << 32);     // lds_addr
  g0.q[1] = (ga & 0x01FFFFFFFFFFFFFFull)             // global_addr[56:0]
          | (2ull << 62);                            // type = image
  unsigned long long fl = (1ull << 16)               // data_size = 2 bytes
                        | (1ull << 20)               // pad_enable
                        | (4ull << 22)               // pad_interval = 32 dwords
                        | (3ull << 25);              // pad_amount   = 4 dwords
  g1.q[0] = fl | (((unsigned long long)(unsigned)tile_w & 0xFFFF) << 48);
  g1.q[1] = (((unsigned long long)(unsigned)tile_w) >> 16)         // dim0 hi
          | (((unsigned long long)1u << 30) << 16)                 // tensor_dim1
          | (((unsigned long long)(unsigned)tile_w) << 48);        // tile_dim0
  g1.q[2] = (unsigned long long)(unsigned)tile_h                   // tile_dim1
          | (((unsigned long long)(unsigned)row_stride_elems) << 32);
  g1.q[3] = 0;
  v4i z4 = {};
#if defined(__clang_major__) && (__clang_major__ >= 23)
  v8i z8 = {};
  __builtin_amdgcn_tensor_load_to_lds(g0.v, g1.v, z4, z4, z8, 0);
#else
  __builtin_amdgcn_tensor_load_to_lds(g0.v, g1.v, z4, z4, 0);
#endif
#else
  (void)lds_dst; (void)gsrc; (void)tile_w; (void)tile_h; (void)row_stride_elems;
#endif
}
__device__ __forceinline__ void wait_tensor() {
#if HAVE_TDM
  __builtin_amdgcn_s_wait_tensorcnt(0);
#endif
}

// ---------------------------------------------------------------------------
// f32 -> bf16 conversion (grid-stride)
// ---------------------------------------------------------------------------
__global__ void convert_bf16_kernel(const float* __restrict__ in,
                                    __bf16* __restrict__ out, size_t n) {
  size_t i = (size_t)blockIdx.x * blockDim.x + threadIdx.x;
  size_t stride = (size_t)gridDim.x * blockDim.x;
  for (; i < n; i += stride) out[i] = (__bf16)in[i];
}

// --- staging helpers -------------------------------------------------------
__device__ __forceinline__ void stage_A_async(int tid, __bf16 (*dst)[40],
                                              const __bf16* src, int ld) {
#pragma unroll
  for (int it = 0; it < 2; ++it) {
    int chunk = tid + it * 128;
    int r = chunk >> 2, c = (chunk & 3) * 8;
    copy16_async(&dst[r][c], &src[(size_t)r * ld + c]);
  }
}
__device__ __forceinline__ void load_W_regs(int tid, const __bf16* src, int ld,
                                            uint4 reg[2]) {
  int r = tid >> 2, c0 = (tid & 3) * 16;
  const uint4* p = reinterpret_cast<const uint4*>(&src[(size_t)r * ld + c0]);
  reg[0] = p[0];
  reg[1] = p[1];
}
__device__ __forceinline__ void store_W_T(int tid, __bf16 (*dst)[40],
                                          const uint4 reg[2]) {
  int r = tid >> 2, c0 = (tid & 3) * 16;
  const __bf16* wp = reinterpret_cast<const __bf16*>(reg);
#pragma unroll
  for (int e = 0; e < 16; ++e) dst[c0 + e][r] = wp[e];
}

// ---------------------------------------------------------------------------
// C = A[M,K](bf16) @ W[K,N](bf16) + bias[N](f32); out bf16 or f32.
// 128 threads = 4 waves; 64x64 tile, BK=32; double-buffered, async A staging,
// W staged transposed so B fragments are contiguous ds_load_b128.
// ---------------------------------------------------------------------------
template <int F32OUT>
__global__ __launch_bounds__(128)
void gemm_bias_kernel(const __bf16* __restrict__ A, const __bf16* __restrict__ W,
                      const float* __restrict__ bias, void* __restrict__ outp,
                      int M, int N, int K) {
  constexpr int BM = 64, BN = 64, BK = 32;
  __shared__ __bf16 At[2][BM][BK + 8];   // row stride 80B
  __shared__ __bf16 WtT[2][BN][BK + 8];  // transposed W tile: [n][k]

  const int tid  = threadIdx.x;
  const int lane = tid & 31, wave = tid >> 5;
  const int half = lane >> 4, ll = lane & 15;
  const int m0 = blockIdx.y * BM, n0 = blockIdx.x * BN;

  const __bf16* Ablk = A + (size_t)m0 * K;

  v8f acc[4];
#pragma unroll
  for (int t = 0; t < 4; ++t) acc[t] = {};

  // Prologue: stage tile k0 = 0 into buffer 0.
  stage_A_async(tid, At[0], Ablk, K);
  {
    uint4 wr[2];
    load_W_regs(tid, W + n0, N, wr);
    store_W_T(tid, WtT[0], wr);
  }
  wait_async();
  __syncthreads();

  int buf = 0;
  for (int k0 = 0; k0 < K; k0 += BK) {
    const bool hasNext = (k0 + BK) < K;
    uint4 wr[2];
    if (hasNext) {
      stage_A_async(tid, At[buf ^ 1], Ablk + (k0 + BK), K);
      load_W_regs(tid, W + (size_t)(k0 + BK) * N + n0, N, wr);
    }

    // A fragment (contiguous 16B runs -> ds_load_b128).
    v16bf a;
    {
      int r = wave * 16 + ll;
#pragma unroll
      for (int j = 0; j < 8; ++j) {
        int kb = kmapA(j, half);
        a[2 * j]     = At[buf][r][kb];
        a[2 * j + 1] = At[buf][r][kb + 1];
      }
    }
#pragma unroll
    for (int t = 0; t < 4; ++t) {
      v16bf b;
      int n = t * 16 + ll;
#pragma unroll
      for (int j = 0; j < 8; ++j) {
        int k = half * 16 + 2 * j;
        b[2 * j]     = WtT[buf][n][k];
        b[2 * j + 1] = WtT[buf][n][k + 1];
      }
      acc[t] = __builtin_amdgcn_wmma_f32_16x16x32_bf16(
          false, a, false, b, (short)0, acc[t], false, false);
    }

    if (hasNext) store_W_T(tid, WtT[buf ^ 1], wr);
    wait_async();
    __syncthreads();
    buf ^= 1;
  }

#pragma unroll
  for (int t = 0; t < 4; ++t) {
    int col = n0 + t * 16 + ll;
    float bv = bias[col];
#pragma unroll
    for (int i = 0; i < 8; ++i) {
      int row = m0 + wave * 16 + half * 8 + i;
      float v = acc[t][i] + bv;
      if (F32OUT)
        ((float*)outp)[(size_t)row * N + col] = v;
      else
        ((__bf16*)outp)[(size_t)row * N + col] = (__bf16)v;
    }
  }
}

// --- attention staging helpers --------------------------------------------
__device__ __forceinline__ void stage_K_tile(int tid, int wave,
                                             __bf16 (*dst)[72],
                                             const __bf16* src) {
#if HAVE_TDM
  (void)tid;
  if (wave == 0) tdm_load_tile_2d(&dst[0][0], src, 64, 64, C_);
#else
  (void)wave;
#pragma unroll
  for (int it = 0; it < 4; ++it) {
    int chunk = tid + it * 128;
    int r = chunk >> 3, c = (chunk & 7) * 8;
    *reinterpret_cast<uint4*>(&dst[r][c]) =
        *reinterpret_cast<const uint4*>(&src[(size_t)r * C_ + c]);
  }
#endif
}
__device__ __forceinline__ void load_V_regs(int tid, const __bf16* src,
                                            uint4 reg[4]) {
  int r = tid >> 1, d0 = (tid & 1) * 32;
  const uint4* p = reinterpret_cast<const uint4*>(&src[(size_t)r * C_ + d0]);
#pragma unroll
  for (int c = 0; c < 4; ++c) reg[c] = p[c];
}
__device__ __forceinline__ void store_V_T(int tid, __bf16 (*dst)[72],
                                          const uint4 reg[4]) {
  int r = tid >> 1, d0 = (tid & 1) * 32;
  const __bf16* vp = reinterpret_cast<const __bf16*>(reg);
#pragma unroll
  for (int e = 0; e < 32; ++e) dst[d0 + e][r] = vp[e];
}

// ---------------------------------------------------------------------------
// Flash attention (causal). Block = 64 query rows of one (b,h); 4 waves x 16.
// K tile via Tensor Data Mover (double-buffered); V tile staged transposed in
// LDS so P@V B-fragments are contiguous ds_load_b128. Softmax reductions via
// DPP16 (no LDS traffic).
// ---------------------------------------------------------------------------
__global__ __launch_bounds__(128)
void attention_kernel(const __bf16* __restrict__ Q, const __bf16* __restrict__ Km,
                      const __bf16* __restrict__ Vm, __bf16* __restrict__ Y) {
  __shared__ __bf16 Ks[2][64][72];
  __shared__ __bf16 VsT[2][64][72];  // transposed: [d][key]
  __shared__ __bf16 Ps[4][16][72];

  const int tid  = threadIdx.x;
  const int lane = tid & 31, wave = tid >> 5;
  const int half = lane >> 4, ll = lane & 15;
  const int b = blockIdx.y / H_, h = blockIdx.y % H_;
  const int q0    = blockIdx.x * 64 + wave * 16;
  const int qmaxb = blockIdx.x * 64 + 63;

  // Preload Q fragments: two 16x32 K-chunks over D=64.
  v16bf aQ[2];
  {
    const __bf16* qp = Q + ((size_t)(b * T_ + q0 + ll)) * C_ + h * D_;
#pragma unroll
    for (int c = 0; c < 2; ++c)
#pragma unroll
      for (int j = 0; j < 8; ++j) {
        int d = c * 32 + kmapA(j, half);
        aQ[c][2 * j]     = qp[d];
        aQ[c][2 * j + 1] = qp[d + 1];
      }
  }

  float rowmax[8], rowsum[8];
  v8f oacc[4];
#pragma unroll
  for (int i = 0; i < 8; ++i) { rowmax[i] = -__builtin_inff(); rowsum[i] = 0.f; }
#pragma unroll
  for (int t = 0; t < 4; ++t) oacc[t] = {};

  const __bf16* kbase = Km + ((size_t)b * S_) * C_ + h * D_;
  const __bf16* vbase = Vm + ((size_t)b * S_) * C_ + h * D_;

  // Prologue: stage tile 0.
  stage_K_tile(tid, wave, Ks[0], kbase);
  {
    uint4 vr[4];
    load_V_regs(tid, vbase, vr);
    store_V_T(tid, VsT[0], vr);
  }
  if (wave == 0) wait_tensor();
  __syncthreads();

  int buf = 0;
  for (int s0 = 0; s0 <= qmaxb; s0 += 64) {
    const bool hasNext = (s0 + 64) <= qmaxb;
    uint4 vr[4];
    if (hasNext) {
      stage_K_tile(tid, wave, Ks[buf ^ 1], kbase + (size_t)(s0 + 64) * C_);
      load_V_regs(tid, vbase + (size_t)(s0 + 64) * C_, vr);
    }

    // Scores: S = Q @ K^T (K rows contiguous in LDS -> b128 fragment loads).
    v8f sc[4];
#pragma unroll
    for (int t = 0; t < 4; ++t) {
      sc[t] = {};
#pragma unroll
      for (int kc = 0; kc < 2; ++kc) {
        v16bf bK;
        int key = t * 16 + ll;
#pragma unroll
        for (int j = 0; j < 8; ++j) {
          int d = kc * 32 + 2 * j + half * 16;
          bK[2 * j]     = Ks[buf][key][d];
          bK[2 * j + 1] = Ks[buf][key][d + 1];
        }
        sc[t] = __builtin_amdgcn_wmma_f32_16x16x32_bf16(
            false, aQ[kc], false, bK, (short)0, sc[t], false, false);
      }
    }

    // Scale (+ causal mask only on diagonal-crossing tiles) + per-row max.
    // Wave-uniform branch: EXEC stays all-ones (WMMA-legal).
    float tmax[8];
#pragma unroll
    for (int i = 0; i < 8; ++i) tmax[i] = -__builtin_inff();
    const bool needMask = (s0 + 63) > q0;
    if (needMask) {
#pragma unroll
      for (int t = 0; t < 4; ++t) {
        int key = s0 + t * 16 + ll;
#pragma unroll
        for (int i = 0; i < 8; ++i) {
          int row = q0 + half * 8 + i;
          float v = sc[t][i] * 0.125f;  // 1/sqrt(64)
          if (key > row) v = -__builtin_inff();
          sc[t][i] = v;
          tmax[i] = fmaxf(tmax[i], v);
        }
      }
    } else {
#pragma unroll
      for (int t = 0; t < 4; ++t) {
#pragma unroll
        for (int i = 0; i < 8; ++i) {
          float v = sc[t][i] * 0.125f;
          sc[t][i] = v;
          tmax[i] = fmaxf(tmax[i], v);
        }
      }
    }
#pragma unroll
    for (int i = 0; i < 8; ++i) tmax[i] = red16_max(tmax[i]);

    float corr[8], psum[8];
#pragma unroll
    for (int i = 0; i < 8; ++i) {
      float mnew = fmaxf(rowmax[i], tmax[i]);
      corr[i]   = __expf(rowmax[i] - mnew);
      rowmax[i] = mnew;
      psum[i]   = 0.f;
    }

    // P = exp(S - m), staged to LDS for C-layout -> A-layout transfer.
#pragma unroll
    for (int t = 0; t < 4; ++t) {
#pragma unroll
      for (int i = 0; i < 8; ++i) {
        float p = __expf(sc[t][i] - rowmax[i]);
        psum[i] += p;
        Ps[wave][half * 8 + i][t * 16 + ll] = (__bf16)p;
      }
    }
#pragma unroll
    for (int i = 0; i < 8; ++i) {
      psum[i] = red16_sum(psum[i]);
      rowsum[i] = rowsum[i] * corr[i] + psum[i];
    }
#pragma unroll
    for (int t = 0; t < 4; ++t)
#pragma unroll
      for (int i = 0; i < 8; ++i) oacc[t][i] *= corr[i];

    // O += P @ V   (V transposed in LDS -> contiguous b128 fragment loads).
#pragma unroll
    for (int kc = 0; kc < 2; ++kc) {
      v16bf aP;
#pragma unroll
      for (int j = 0; j < 8; ++j) {
        int k = kc * 32 + kmapA(j, half);
        aP[2 * j]     = Ps[wave][ll][k];
        aP[2 * j + 1] = Ps[wave][ll][k + 1];
      }
#pragma unroll
      for (int t = 0; t < 4; ++t) {
        v16bf bV;
        int d = t * 16 + ll;
#pragma unroll
        for (int j = 0; j < 8; ++j) {
          int k = kc * 32 + half * 16 + 2 * j;
          bV[2 * j]     = VsT[buf][d][k];
          bV[2 * j + 1] = VsT[buf][d][k + 1];
        }
        oacc[t] = __builtin_amdgcn_wmma_f32_16x16x32_bf16(
            false, aP, false, bV, (short)0, oacc[t], false, false);
      }
    }

    if (hasNext) store_V_T(tid, VsT[buf ^ 1], vr);
    if (wave == 0) wait_tensor();
    __syncthreads();
    buf ^= 1;
  }

  // Normalize and write Y (bf16, [B*T, C] with head at col h*D).
#pragma unroll
  for (int t = 0; t < 4; ++t) {
#pragma unroll
    for (int i = 0; i < 8; ++i) {
      int row = q0 + half * 8 + i;
      int d = t * 16 + ll;
      float v = oacc[t][i] / rowsum[i];
      Y[((size_t)(b * T_ + row)) * C_ + h * D_ + d] = (__bf16)v;
    }
  }
}

// ---------------------------------------------------------------------------
extern "C" void kernel_launch(void* const* d_in, const int* in_sizes, int n_in,
                              void* d_out, int out_size, void* d_ws, size_t ws_size,
                              hipStream_t stream) {
  const float* x   = (const float*)d_in[0];
  const float* ctx = (const float*)d_in[1];
  const float* q_w = (const float*)d_in[2];
  const float* q_b = (const float*)d_in[3];
  const float* k_w = (const float*)d_in[4];
  const float* k_b = (const float*)d_in[5];
  const float* v_w = (const float*)d_in[6];
  const float* v_b = (const float*)d_in[7];
  const float* o_w = (const float*)d_in[8];
  const float* o_b = (const float*)d_in[9];
  (void)in_sizes; (void)n_in; (void)out_size; (void)ws_size;

  const size_t BT  = (size_t)B_ * T_;     // 4096
  const size_t tok = BT * C_;             // token-matrix elements
  const size_t wsz = (size_t)C_ * C_;     // weight elements

  char* ws = (char*)d_ws;
  size_t off = 0;
  auto alloc = [&](size_t elems) {
    __bf16* p = (__bf16*)(ws + off);
    off += ((elems * 2 + 255) & ~(size_t)255);
    return p;
  };
  __bf16* xb  = alloc(tok);
  __bf16* cb  = alloc(tok);
  __bf16* qwb = alloc(wsz);
  __bf16* kwb = alloc(wsz);
  __bf16* vwb = alloc(wsz);
  __bf16* owb = alloc(wsz);
  __bf16* qb  = alloc(tok);
  __bf16* kb  = alloc(tok);
  __bf16* vb  = alloc(tok);
  __bf16* yb  = alloc(tok);

  convert_bf16_kernel<<<1024, 256, 0, stream>>>(x,   xb, tok);
  convert_bf16_kernel<<<1024, 256, 0, stream>>>(ctx, cb, tok);
  convert_bf16_kernel<<<512,  256, 0, stream>>>(q_w, qwb, wsz);
  convert_bf16_kernel<<<512,  256, 0, stream>>>(k_w, kwb, wsz);
  convert_bf16_kernel<<<512,  256, 0, stream>>>(v_w, vwb, wsz);
  convert_bf16_kernel<<<512,  256, 0, stream>>>(o_w, owb, wsz);

  dim3 gP(C_ / 64, BT / 64);
  gemm_bias_kernel<0><<<gP, 128, 0, stream>>>(xb, qwb, q_b, qb, (int)BT, C_, C_);
  gemm_bias_kernel<0><<<gP, 128, 0, stream>>>(cb, kwb, k_b, kb, (int)BT, C_, C_);
  gemm_bias_kernel<0><<<gP, 128, 0, stream>>>(cb, vwb, v_b, vb, (int)BT, C_, C_);

  attention_kernel<<<dim3(T_ / 64, B_ * H_), 128, 0, stream>>>(qb, kb, vb, yb);

  gemm_bias_kernel<1><<<gP, 128, 0, stream>>>(yb, owb, o_b, d_out, (int)BT, C_, C_);
}
